// Seq2SeqAttention_56075093016914
// MI455X (gfx1250) — matrix operand order repaired and checked by reading, not compiled, not run on hardware
//
#include <hip/hip_runtime.h>
#include <hip/hip_bf16.h>

// ---------------------------------------------------------------------------
// Seq2Seq LSTM + attention for MI455X (gfx1250, wave32, WMMA 16x16x32 f16)
//
//   1. weights -> f16 in workspace (L2-resident; sm_w f16 = 8 MB << 192 MB L2)
//   2. gather embeddings -> f16 activation matrices (time-major rows)
//   3. batched WMMA GEMMs for non-recurrent x@Wih parts (all timesteps),
//      8-wave blocks with async-staged (ASYNCcnt) B panel in LDS
//   4. persistent single-WGP encoder: 50 LSTM steps; Whh async-staged to LDS
//      once; h@Whh via v_wmma_f32_16x16x32_f16 from LDS
//   5. batched WMMA GEMM: att_src = enc_out @ att_w1s^T
//   6. persistent single-WGP decoder: 49 steps of cell + attention + final
//   7. ONE WMMA GEMM (1568x32000, K=128) for all logits, non-temporal stores,
//      writing the (B, T-1, V) permuted layout directly.
// ---------------------------------------------------------------------------

typedef _Float16 f16;
typedef _Float16 v16h __attribute__((ext_vector_type(16)));
typedef _Float16 h8   __attribute__((ext_vector_type(8)));
typedef float    v8f  __attribute__((ext_vector_type(8)));

__device__ __forceinline__ float sigf(float x) { return 1.0f / (1.0f + __expf(-x)); }

// ---- WMMA fragment helpers (layouts per CDNA5 ISA 7.12.2) -------

// A fragment: 16x32 f16 tile at (m0,k0) of row-major A, leading dim lda.
// Lane (row = lane&15, half = lane>>4) holds row m0+row:
//   elements 0..7  -> K = k0 + 8*half + 0..7
//   elements 8..15 -> K = k0 + 16 + 8*half + 0..7
__device__ __forceinline__ v16h load_a_frag(const f16* A, int lda, int m0, int k0,
                                            int row, int half) {
  union { v16h v; h8 h[2]; } u;
  const f16* p = A + (long long)(m0 + row) * lda + k0 + 8 * half;
  u.h[0] = *(const h8*)(p);
  u.h[1] = *(const h8*)(p + 16);
  return u.v;
}

// B fragment for C = A @ W^T: B column n == W row n.
// Lane (row, half) holds W row n0+row, K = k0 + 8*half + 0..7, k0 + 16 + 8*half + 0..7.
__device__ __forceinline__ v16h load_b_frag(const f16* W, int ldw, int n0, int k0,
                                            int row, int half) {
  union { v16h v; h8 h[2]; } u;
  const f16* p = W + (long long)(n0 + row) * ldw + k0 + 8 * half;
  u.h[0] = *(const h8*)(p);
  u.h[1] = *(const h8*)(p + 16);
  return u.v;
}

__device__ __forceinline__ v8f wmma16(v16h a, v16h b, v8f c) {
  return __builtin_amdgcn_wmma_f32_16x16x32_f16(false, a, false, b, (short)0, c,
                                                false, false);
}

// Async DMA: copy 16 bytes from global to LDS (byte offset into dynamic LDS).
// gfx1250 ASYNCcnt path (CDNA5 ISA opcode 98 / CDNA5 ISA).
__device__ __forceinline__ void async_g2l_b128(unsigned ldsOff, const void* gptr) {
  asm volatile("global_load_async_to_lds_b128 %0, %1, off"
               :: "v"(ldsOff), "v"(gptr) : "memory");
}
__device__ __forceinline__ void wait_asynccnt0() {
  asm volatile("s_wait_asynccnt 0x0" ::: "memory");
}

// ---- small utility kernels --------------------------------------------------

__global__ void k_cvt_f16(const float* __restrict__ s, f16* __restrict__ d, int n) {
  int i = blockIdx.x * blockDim.x + threadIdx.x;
  if (i < n) d[i] = (f16)s[i];
}

// dst rows are time-major: row m = step*32 + b ; E = 128
__global__ void k_gather(const int* __restrict__ tok, int tokLd, int nsteps,
                         const float* __restrict__ emb, f16* __restrict__ dst) {
  int idx = blockIdx.x * blockDim.x + threadIdx.x;
  int total = nsteps * 32 * 128;
  if (idx >= total) return;
  int e = idx & 127;
  int m = idx >> 7;      // s*32 + b
  int b = m & 31;
  int s = m >> 5;
  int t = tok[b * tokLd + s];
  dst[idx] = (f16)emb[(long long)t * 128 + e];
}

// ---- batched WMMA GEMM:  C = A(MxK,f16) @ W(NxK,f16)^T + bias --------------
// 256 threads (8 wave32s). Block tile = 128(M) x 64(N). The 64xK B panel is
// async-staged into LDS once (shared by all 8 M-tiles -> 8x less L2 traffic),
// then each wave computes a 16x64 strip, A-frag reused across 4 B tiles.
// Output mapping: C[(m % MB)*sB + (m / MB)*sT + n]  (handles (B,T,V) permute).
// nt != 0 -> non-temporal stores (streaming output, keep L2 for weights).
__global__ void k_gemm(const f16* __restrict__ A, const f16* __restrict__ W,
                       const float* __restrict__ bias0, const float* __restrict__ bias1,
                       float* __restrict__ C, int M, int N, int K,
                       int MB, long long sB, long long sT, int nt) {
  extern __shared__ char smem[];
  f16* Bs = (f16*)smem;                  // 64 x K halves
  int tid = threadIdx.x;
  int wave = tid >> 5, lane = tid & 31;
  int row = lane & 15, half = lane >> 4;
  int n0 = blockIdx.x * 64;
  int m0 = blockIdx.y * 128 + wave * 16;

  // cooperative async stage of B panel: rows n0..n0+63, all K
  int chunks = (64 * K) >> 3;            // 16-byte chunks
  for (int c = tid; c < chunks; c += 256) {
    int r   = (c * 8) / K;
    int col = (c * 8) % K;
    async_g2l_b128((unsigned)(c * 16), W + (long long)(n0 + r) * K + col);
  }
  wait_asynccnt0();
  __syncthreads();

  if (m0 < M) {
    v8f acc[4] = {};
    for (int k0 = 0; k0 < K; k0 += 32) {
      v16h a = load_a_frag(A, K, m0, k0, row, half);
#pragma unroll
      for (int i = 0; i < 4; i++) {
        v16h b = load_b_frag(Bs, K, 16 * i, k0, row, half);
        acc[i] = wmma16(a, b, acc[i]);
      }
    }
#pragma unroll
    for (int i = 0; i < 4; i++) {
      int n = n0 + 16 * i + row;
      float bv = (bias0 ? bias0[n] : 0.0f) + (bias1 ? bias1[n] : 0.0f);
#pragma unroll
      for (int j = 0; j < 8; j++) {
        int m = m0 + j + 8 * half;
        float* dst = C + (long long)(m % MB) * sB + (long long)(m / MB) * sT + n;
        float v = acc[i][j] + bv;
        if (nt) __builtin_nontemporal_store(v, dst);
        else    *dst = v;
      }
    }
  }
}

// ---- persistent encoder: 50 LSTM steps, one workgroup (16 wave32s) ---------
// LDS: h16(8K) cbuf(16K) gates(64K) Whh16(128K) = 216 KB (< 320 KB/WGP)
__global__ void k_enc_cell(const float* __restrict__ Gx, const f16* __restrict__ Whh,
                           float* __restrict__ enc_out, f16* __restrict__ enc_out16,
                           int S_) {
  extern __shared__ char smem[];
  f16*   h16   = (f16*)smem;                       // 32*128 f16  @0
  float* cbuf  = (float*)(smem + 8192);            // 32*128 f32  @8192
  float* gates = (float*)(smem + 24576);           // 32*512 f32  @24576
  f16*   whhL  = (f16*)(smem + 90112);             // 512*128 f16 @90112
  int tid = threadIdx.x;                 // 0..511
  int wave = tid >> 5, lane = tid & 31;
  int row = lane & 15, half = lane >> 4;

  // stage Whh (128 KB) into LDS once via async DMA
  for (int c = tid; c < (512 * 128) / 8; c += 512)
    async_g2l_b128((unsigned)(90112 + c * 16), Whh + c * 8);
  for (int r = 0; r < 8; r++) { int i = tid + r * 512; h16[i] = (f16)0.0f; cbuf[i] = 0.0f; }
  wait_asynccnt0();
  __syncthreads();

  for (int s = 0; s < S_; s++) {
    // h @ Whh^T : 2 M-tiles x 32 N-tiles = 64 tiles, 4 per wave (LDS-fed WMMA)
#pragma unroll
    for (int ti = 0; ti < 4; ti++) {
      int tau = wave * 4 + ti;
      int m0 = (tau & 1) * 16;
      int n0 = (tau >> 1) * 16;
      v8f acc = {};
#pragma unroll
      for (int k0 = 0; k0 < 128; k0 += 32) {
        v16h a = load_a_frag(h16, 128, m0, k0, row, half);
        v16h b = load_b_frag(whhL, 128, n0, k0, row, half);
        acc = wmma16(a, b, acc);
      }
      int n = n0 + row;
#pragma unroll
      for (int j = 0; j < 8; j++) {
        int m = m0 + j + 8 * half;
        gates[m * 512 + n] = acc[j] + Gx[((long long)s * 32 + m) * 512 + n];
      }
    }
    __syncthreads();
#pragma unroll
    for (int r = 0; r < 8; r++) {
      int idx = tid + r * 512;
      int b = idx >> 7, j = idx & 127;
      float ig = gates[b * 512 + j];
      float fg = gates[b * 512 + 128 + j];
      float gg = gates[b * 512 + 256 + j];
      float og = gates[b * 512 + 384 + j];
      float c  = cbuf[idx];
      float cn = sigf(fg) * c + sigf(ig) * tanhf(gg);
      float hn = sigf(og) * tanhf(cn);
      cbuf[idx] = cn;
      h16[idx]  = (f16)hn;
      long long o = ((long long)s * 32 + b) * 128 + j;
      enc_out[o]   = hn;
      enc_out16[o] = (f16)hn;
    }
    __syncthreads();
  }
}

// ---- persistent decoder: 49 steps of cell + attention + final --------------
// LDS: h16(8K) cbuf(16K) gates(64K) atgt(16K) sc(8K) cat16(16K) Whh(128K) = 256 KB
__global__ void k_dec_cell(const float* __restrict__ Gx, const f16* __restrict__ Whh,
                           const float* __restrict__ att_src, const f16* __restrict__ w1t16,
                           const float* __restrict__ b1t, const float* __restrict__ w2,
                           const float* __restrict__ b2, const float* __restrict__ enc_out,
                           const f16* __restrict__ outw16, const float* __restrict__ outb,
                           f16* __restrict__ final16, int T_, int S_) {
  extern __shared__ char smem[];
  f16*   h16   = (f16*)smem;                        // 32*128 f16  @0
  float* cbuf  = (float*)(smem + 8192);             // 32*128 f32  @8192
  float* gates = (float*)(smem + 24576);            // 32*512 f32  @24576
  float* atgt  = (float*)(smem + 90112);            // 32*128 f32  @90112
  float* sc    = (float*)(smem + 106496);           // 32*64  f32  @106496
  f16*   cat16 = (f16*)(smem + 114688);             // 32*256 f16  @114688
  f16*   whhL  = (f16*)(smem + 131072);             // 512*128 f16 @131072
  int tid = threadIdx.x;
  int wave = tid >> 5, lane = tid & 31;
  int row = lane & 15, half = lane >> 4;

  for (int c = tid; c < (512 * 128) / 8; c += 512)
    async_g2l_b128((unsigned)(131072 + c * 16), Whh + c * 8);
  for (int r = 0; r < 8; r++) { int i = tid + r * 512; h16[i] = (f16)0.0f; cbuf[i] = 0.0f; }
  wait_asynccnt0();
  __syncthreads();

  for (int t = 0; t < T_; t++) {
    // ---- gates = Gx[t] + h @ Whh^T (WMMA from LDS) ----
#pragma unroll
    for (int ti = 0; ti < 4; ti++) {
      int tau = wave * 4 + ti;
      int m0 = (tau & 1) * 16;
      int n0 = (tau >> 1) * 16;
      v8f acc = {};
#pragma unroll
      for (int k0 = 0; k0 < 128; k0 += 32) {
        v16h a = load_a_frag(h16, 128, m0, k0, row, half);
        v16h b = load_b_frag(whhL, 128, n0, k0, row, half);
        acc = wmma16(a, b, acc);
      }
      int n = n0 + row;
#pragma unroll
      for (int j = 0; j < 8; j++) {
        int m = m0 + j + 8 * half;
        gates[m * 512 + n] = acc[j] + Gx[((long long)t * 32 + m) * 512 + n];
      }
    }
    __syncthreads();
    // ---- LSTM cell update; fill first half of cat16 with h ----
#pragma unroll
    for (int r = 0; r < 8; r++) {
      int idx = tid + r * 512;
      int b = idx >> 7, j = idx & 127;
      float ig = gates[b * 512 + j];
      float fg = gates[b * 512 + 128 + j];
      float gg = gates[b * 512 + 256 + j];
      float og = gates[b * 512 + 384 + j];
      float c  = cbuf[idx];
      float cn = sigf(fg) * c + sigf(ig) * tanhf(gg);
      float hn = sigf(og) * tanhf(cn);
      cbuf[idx] = cn;
      h16[idx]  = (f16)hn;
      cat16[b * 256 + j] = (f16)hn;
    }
    __syncthreads();
    // ---- att_tgt = h @ w1t^T + b1t : 16 tiles, one per wave (WMMA) ----
    {
      int m0 = (wave & 1) * 16;
      int n0 = (wave >> 1) * 16;
      v8f acc = {};
#pragma unroll
      for (int k0 = 0; k0 < 128; k0 += 32) {
        v16h a = load_a_frag(h16, 128, m0, k0, row, half);
        v16h b = load_b_frag(w1t16, 128, n0, k0, row, half);
        acc = wmma16(a, b, acc);
      }
      int n = n0 + row;
#pragma unroll
      for (int j = 0; j < 8; j++) {
        int m = m0 + j + 8 * half;
        atgt[m * 128 + n] = acc[j] + b1t[n];
      }
    }
    __syncthreads();
    // ---- scores[b][s] = sum_a tanh(att_src + att_tgt) * w2[a] + b2 ----
    for (int p = tid; p < 32 * S_; p += 512) {
      int s = p >> 5, b = p & 31;
      const float* as = att_src + (long long)p * 128;   // row s*32+b
      const float* at = atgt + b * 128;
      float acc = 0.0f;
      for (int a = 0; a < 128; a++) acc += tanhf(as[a] + at[a]) * w2[a];
      sc[b * 64 + s] = acc + b2[0];
    }
    __syncthreads();
    // ---- softmax over s (one thread per batch row) ----
    if (tid < 32) {
      int b = tid;
      float mx = -1e30f;
      for (int s = 0; s < S_; s++) mx = fmaxf(mx, sc[b * 64 + s]);
      float sum = 0.0f;
      for (int s = 0; s < S_; s++) { float e = __expf(sc[b * 64 + s] - mx); sc[b * 64 + s] = e; sum += e; }
      float inv = 1.0f / sum;
      for (int s = 0; s < S_; s++) sc[b * 64 + s] *= inv;
    }
    __syncthreads();
    // ---- att_vec[b][hh] = sum_s align * enc_out ; second half of cat16 ----
#pragma unroll
    for (int r = 0; r < 8; r++) {
      int idx = tid + r * 512;
      int b = idx >> 7, hh = idx & 127;
      float acc = 0.0f;
      for (int s = 0; s < S_; s++)
        acc += sc[b * 64 + s] * enc_out[((long long)s * 32 + b) * 128 + hh];
      cat16[b * 256 + 128 + hh] = (f16)acc;
    }
    __syncthreads();
    // ---- final = tanh(cat @ out_w^T + out_b) : 16 tiles, K=256 (WMMA) ----
    {
      int m0 = (wave & 1) * 16;
      int n0 = (wave >> 1) * 16;
      v8f acc = {};
#pragma unroll
      for (int k0 = 0; k0 < 256; k0 += 32) {
        v16h a = load_a_frag(cat16, 256, m0, k0, row, half);
        v16h b = load_b_frag(outw16, 256, n0, k0, row, half);
        acc = wmma16(a, b, acc);
      }
      int n = n0 + row;
#pragma unroll
      for (int j = 0; j < 8; j++) {
        int m = m0 + j + 8 * half;
        float f = tanhf(acc[j] + outb[n]);
        final16[((long long)t * 32 + m) * 128 + n] = (f16)f;
      }
    }
    __syncthreads();
  }
}

// ---------------------------------------------------------------------------

extern "C" void kernel_launch(void* const* d_in, const int* in_sizes, int n_in,
                              void* d_out, int out_size, void* d_ws, size_t ws_size,
                              hipStream_t stream) {
  (void)in_sizes; (void)n_in; (void)out_size; (void)ws_size;
  const int*   src     = (const int*)d_in[0];
  const int*   trg     = (const int*)d_in[1];
  const float* emb_src = (const float*)d_in[2];
  const float* emb_trg = (const float*)d_in[3];
  const float* enc_Wih = (const float*)d_in[4];
  const float* enc_Whh = (const float*)d_in[5];
  const float* enc_bih = (const float*)d_in[6];
  const float* enc_bhh = (const float*)d_in[7];
  const float* dec_Wih = (const float*)d_in[8];
  const float* dec_Whh = (const float*)d_in[9];
  const float* dec_bih = (const float*)d_in[10];
  const float* dec_bhh = (const float*)d_in[11];
  const float* att_w1s = (const float*)d_in[12];
  const float* att_b1s = (const float*)d_in[13];
  const float* att_w1t = (const float*)d_in[14];
  const float* att_b1t = (const float*)d_in[15];
  const float* att_w2  = (const float*)d_in[16];
  const float* att_b2  = (const float*)d_in[17];
  const float* out_w   = (const float*)d_in[18];
  const float* out_b   = (const float*)d_in[19];
  const float* sm_w    = (const float*)d_in[20];
  const float* sm_b    = (const float*)d_in[21];
  float* out = (float*)d_out;

  const int B = 32, S = 50, T = 50, E = 128, H = 128, V = 32000;
  const int Td = T - 1;                 // 49 decoder steps
  const int Me = B * S;                 // 1600
  const int Md = B * Td;                // 1568

  char* ws = (char*)d_ws;
  size_t o = 0;
  auto alloc = [&](size_t bytes) { size_t r = o; o = (o + bytes + 255) & ~(size_t)255; return r; };
  f16*   encWih16 = (f16*)(ws + alloc((size_t)4 * H * E * 2));
  f16*   encWhh16 = (f16*)(ws + alloc((size_t)4 * H * H * 2));
  f16*   decWih16 = (f16*)(ws + alloc((size_t)4 * H * E * 2));
  f16*   decWhh16 = (f16*)(ws + alloc((size_t)4 * H * H * 2));
  f16*   w1s16    = (f16*)(ws + alloc((size_t)128 * H * 2));
  f16*   w1t16    = (f16*)(ws + alloc((size_t)128 * H * 2));
  f16*   outw16   = (f16*)(ws + alloc((size_t)H * 2 * H * 2));
  f16*   smw16    = (f16*)(ws + alloc((size_t)V * H * 2));
  f16*   Xsrc16   = (f16*)(ws + alloc((size_t)Me * E * 2));
  f16*   Xtrg16   = (f16*)(ws + alloc((size_t)Md * E * 2));
  float* GxE      = (float*)(ws + alloc((size_t)Me * 512 * 4));
  float* GxD      = (float*)(ws + alloc((size_t)Md * 512 * 4));
  float* encOut   = (float*)(ws + alloc((size_t)Me * H * 4));
  f16*   encOut16 = (f16*)(ws + alloc((size_t)Me * H * 2));
  float* attSrc   = (float*)(ws + alloc((size_t)Me * 128 * 4));
  f16*   fin16    = (f16*)(ws + alloc((size_t)Md * H * 2));

  auto cvt = [&](const float* s, f16* d, int n) {
    k_cvt_f16<<<(n + 255) / 256, 256, 0, stream>>>(s, d, n);
  };
  cvt(enc_Wih, encWih16, 4 * H * E);
  cvt(enc_Whh, encWhh16, 4 * H * H);
  cvt(dec_Wih, decWih16, 4 * H * E);
  cvt(dec_Whh, decWhh16, 4 * H * H);
  cvt(att_w1s, w1s16, 128 * H);
  cvt(att_w1t, w1t16, 128 * H);
  cvt(out_w, outw16, H * 2 * H);
  cvt(sm_w, smw16, V * H);

  k_gather<<<(Me * E + 255) / 256, 256, 0, stream>>>(src, S, S, emb_src, Xsrc16);
  k_gather<<<(Md * E + 255) / 256, 256, 0, stream>>>(trg, T, Td, emb_trg, Xtrg16);

  const int BPANEL = 64 * 128 * 2;      // LDS bytes for staged B panel (K=128)

  // Gx = X @ Wih^T + bih + bhh   (all timesteps batched)
  k_gemm<<<dim3(512 / 64, (Me + 127) / 128), 256, BPANEL, stream>>>(
      Xsrc16, encWih16, enc_bih, enc_bhh, GxE, Me, 512, E, Me, 512, 0, 0);
  k_gemm<<<dim3(512 / 64, (Md + 127) / 128), 256, BPANEL, stream>>>(
      Xtrg16, decWih16, dec_bih, dec_bhh, GxD, Md, 512, E, Md, 512, 0, 0);

  // encoder: persistent single workgroup, 216 KB dynamic LDS
  k_enc_cell<<<1, 512, 90112 + 131072, stream>>>(GxE, encWhh16, encOut, encOut16, S);

  // att_src = enc_out @ w1s^T + b1s
  k_gemm<<<dim3(128 / 64, (Me + 127) / 128), 256, BPANEL, stream>>>(
      encOut16, w1s16, att_b1s, nullptr, attSrc, Me, 128, H, Me, 128, 0, 0);

  // decoder: persistent single workgroup, 256 KB dynamic LDS
  k_dec_cell<<<1, 512, 131072 + 131072, stream>>>(GxD, decWhh16, attSrc, w1t16,
                                                  att_b1t, att_w2, att_b2, encOut,
                                                  outw16, out_b, fin16, Td, S);

  // logits = final @ sm_w^T + sm_b  -> out[b*(49*V) + t*V + n], NT stores
  k_gemm<<<dim3(V / 64, (Md + 127) / 128), 256, BPANEL, stream>>>(
      fin16, smw16, sm_b, nullptr, out, Md, V, H,
      32, (long long)Td * V, (long long)V, 1);
}
